// PAM1_90632399880208
// MI455X (gfx1250) — matrix operand order repaired and checked by reading, not compiled, not run on hardware
//
#include <hip/hip_runtime.h>
#include <hip/hip_bf16.h>

// ---------------------------------------------------------------------------
// Problem constants (B,H,W,C) = (8,64,64,256); HW=4096; S=B*HW=32768.
// ---------------------------------------------------------------------------
#define BATCH 8
#define HW    4096
#define CH    256
#define ROWS  (BATCH * HW)       // 32768

typedef __attribute__((ext_vector_type(16))) __bf16     v16bf;
typedef __attribute__((ext_vector_type(2)))  __bf16     v2bf;
typedef __attribute__((ext_vector_type(8)))  float      v8f;
typedef __attribute__((ext_vector_type(4)))  unsigned   u32x4;
typedef __attribute__((ext_vector_type(4)))  int        i32x4;

union BFrag { u32x4 u[2]; unsigned w[8]; v16bf b; };

__device__ inline unsigned short f2b(float f) {
    unsigned u = __builtin_bit_cast(unsigned, f);
    return (unsigned short)((u + 0x7FFFu + ((u >> 16) & 1u)) >> 16);  // RNE
}
__device__ inline unsigned pack2(float lo, float hi) {
#if __has_builtin(__builtin_amdgcn_cvt_pk_bf16_f32)
    v2bf p = __builtin_amdgcn_cvt_pk_bf16_f32(lo, hi);
    return __builtin_bit_cast(unsigned, p);
#else
    return (unsigned)f2b(lo) | ((unsigned)f2b(hi) << 16);
#endif
}

// ---------------------------------------------------------------------------
// Async global->LDS copy (CDNA5 ASYNCcnt path) with synchronous fallback.
// Probe-confirmed: param0 is an int4 pointer in the global (__device__) AS.
// ---------------------------------------------------------------------------
#if __has_builtin(__builtin_amdgcn_global_load_async_to_lds_b128)
#define HAVE_ASYNC_LDS 1
#else
#define HAVE_ASYNC_LDS 0
#endif

__device__ inline void copy16(const unsigned short* g, unsigned short* l) {
#if HAVE_ASYNC_LDS
    __builtin_amdgcn_global_load_async_to_lds_b128(
        (__attribute__((address_space(1))) i32x4*)(unsigned long long)g,
        (__attribute__((address_space(3))) i32x4*)(unsigned)(unsigned long long)l,
        0, 0);
#else
    *(u32x4*)l = *(const u32x4*)g;
#endif
}

__device__ inline void wait_async_all() {
#if HAVE_ASYNC_LDS
#if __has_builtin(__builtin_amdgcn_s_wait_asynccnt)
    __builtin_amdgcn_s_wait_asynccnt(0);
#else
    asm volatile("s_wait_asynccnt 0x0" ::: "memory");
#endif
#endif
}

// ---------------------------------------------------------------------------
// Kernel 0: transpose-convert pointwise weights to bf16 [out_c][in_c].
// ---------------------------------------------------------------------------
__global__ void prep_w_kernel(const float* __restrict__ pw1,
                              const float* __restrict__ pw2,
                              unsigned short* __restrict__ w1T,
                              unsigned short* __restrict__ w2T) {
    int n = blockIdx.x;        // out channel
    int k = threadIdx.x;       // in channel
    w1T[n * CH + k] = f2b(pw1[k * CH + n]);
    w2T[n * CH + k] = f2b(pw2[k * CH + n]);
}

// ---------------------------------------------------------------------------
// Kernel 1: projections.  q -> row-major bf16 [row][ch] (A/B operand friendly)
//                         v -> TRANSPOSED bf16 [b][ch][key] (PV B-operand friendly)
// 128 threads = 4 waves; each wave does 16 rows x 256 cols, K-loop over 256.
// ---------------------------------------------------------------------------
__global__ __launch_bounds__(128) void proj_kernel(
        const float* __restrict__ x,
        const float* __restrict__ dw1, const float* __restrict__ b1,
        const float* __restrict__ dw2, const float* __restrict__ b2,
        const unsigned short* __restrict__ w1T,
        const unsigned short* __restrict__ w2T,
        unsigned short* __restrict__ qb,
        unsigned short* __restrict__ vbT) {
    __shared__ unsigned short As[64 * CH];        // 32 KB staged bf16 activations

    const int tid  = threadIdx.x;
    const int wave = tid >> 5;
    const int lane = tid & 31;
    const int half = lane >> 4;
    const int l15  = lane & 15;
    const int rowBase = blockIdx.x * 64;

    for (int pass = 0; pass < 2; ++pass) {
        const float*          dw   = pass ? dw2 : dw1;
        const float*          bias = pass ? b2  : b1;
        const unsigned short* wT   = pass ? w2T : w1T;

        __syncthreads();
        // Stage bf16(x * dw) for 64 rows, 4 elems per op.
        for (int i = tid * 4; i < 64 * CH; i += 128 * 4) {
            int r = i >> 8, c = i & 255;
            float4 xv = *(const float4*)(x + (size_t)(rowBase + r) * CH + c);
            float4 dv = *(const float4*)(dw + c);
            unsigned o0 = pack2(xv.x * dv.x, xv.y * dv.y);
            unsigned o1 = pack2(xv.z * dv.z, xv.w * dv.w);
            *(uint2*)&As[i] = make_uint2(o0, o1);
        }
        __syncthreads();

        const int rw = wave * 16;
        for (int nt = 0; nt < 16; ++nt) {
            v8f acc = {};
#pragma unroll
            for (int ks = 0; ks < 8; ++ks) {
                BFrag a;
                const unsigned short* ap = &As[(rw + l15) * CH + ks * 32 + half * 8];
                a.u[0] = *(const u32x4*)ap;
                a.u[1] = *(const u32x4*)(ap + 16);
                BFrag w;
                const unsigned short* bp = &wT[(nt * 16 + l15) * CH + ks * 32 + half * 8];
                w.u[0] = *(const u32x4*)bp;
                w.u[1] = *(const u32x4*)(bp + 16);
                acc = __builtin_amdgcn_wmma_f32_16x16x32_bf16(
                        false, a.b, false, w.b, (short)0, acc, false, false);
            }
            int   ch = nt * 16 + l15;
            float bv = bias[ch];
            if (pass == 0) {
                // q: row-major scatter (keys strided)
#pragma unroll
                for (int r = 0; r < 8; ++r) {
                    int row = rowBase + rw + r + 8 * half;
                    qb[(size_t)row * CH + ch] = f2b(acc[r] + bv);
                }
            } else {
                // v: transposed store; a lane's 8 rows are 8 contiguous keys
                int row0 = rowBase + rw + 8 * half;
                int bb   = row0 >> 12;            // batch
                int key0 = row0 & (HW - 1);
                BFrag pk;
#pragma unroll
                for (int j = 0; j < 4; ++j)
                    pk.w[j] = pack2(acc[2 * j] + bv, acc[2 * j + 1] + bv);
                *(u32x4*)&vbT[((size_t)bb * CH + ch) * HW + key0] = pk.u[0];
            }
        }
    }
}

// ---------------------------------------------------------------------------
// Kernel 2: flash attention  out = softmax(q q^T) v * gamma + x.
// 512 blocks (8 batches x 64 row-blocks), 128 threads = 4 waves x 16 queries.
// S^T = K Q^T so P converts to a WMMA A-fragment with in-lane bf16 packing.
// K/V tiles double-buffered and staged with async global->LDS copies.
// ---------------------------------------------------------------------------
#define VPITCH 40   // V^T LDS pitch in ushorts (80B: 16B-aligned, conflict-skewed)

__global__ __launch_bounds__(128) void attn_kernel(
        const unsigned short* __restrict__ qb,
        const unsigned short* __restrict__ vbT,
        const float* __restrict__ x,
        const float* __restrict__ gamma,
        float* __restrict__ out) {
    __shared__ unsigned short Qs[64 * CH];            // 32 KB  query tile
    __shared__ unsigned short Ks2[2][32 * CH];        // 2x16 KB key tiles
    __shared__ unsigned short Vts2[2][CH * VPITCH];   // 2x20 KB V tiles [ch][key]

    const int tid  = threadIdx.x;
    const int wave = tid >> 5;
    const int lane = tid & 31;
    const int half = lane >> 4;
    const int l15  = lane & 15;
    const int b    = blockIdx.x >> 6;
    const int mb   = (blockIdx.x & 63) * 64;            // query row base
    const size_t batchOff  = (size_t)b * HW * CH;       // qb / x / out
    const size_t batchOffT = (size_t)b * CH * HW;       // vbT

    // Stage this block's queries (async; covered by first pipeline wait).
    for (int i = tid * 8; i < 64 * CH; i += 128 * 8)
        copy16(&qb[batchOff + (size_t)mb * CH + i], &Qs[i]);

    // Pipeline prologue: stage tile 0 into buffer 0.
    {
        unsigned short* Ks  = Ks2[0];
        unsigned short* Vts = Vts2[0];
        for (int i = tid * 8; i < 32 * CH; i += 128 * 8)
            copy16(&qb[batchOff + i], &Ks[i]);
        for (int i = tid * 8; i < CH * 32; i += 128 * 8) {
            int ch = i >> 5, key = i & 31;
            copy16(&vbT[batchOffT + (size_t)ch * HW + key], &Vts[ch * VPITCH + key]);
        }
    }

    float m = -__builtin_inff();    // running max for query l15
    float l = 0.0f;                 // running denom for query l15
    v8f   acc[16];
#pragma unroll
    for (int nt = 0; nt < 16; ++nt) acc[nt] = (v8f){};

    for (int kb = 0; kb < HW; kb += 32) {
        const int cur = (kb >> 5) & 1;
        wait_async_all();           // this wave's staged tile is in LDS
        __syncthreads();            // everyone's tile (and Qs) visible

        // Overlap: issue async staging of the NEXT tile into the other buffer.
        if (kb + 32 < HW) {
            unsigned short* Kn  = Ks2[cur ^ 1];
            unsigned short* Vn  = Vts2[cur ^ 1];
            for (int i = tid * 8; i < 32 * CH; i += 128 * 8)
                copy16(&qb[batchOff + (size_t)(kb + 32) * CH + i], &Kn[i]);
            for (int i = tid * 8; i < CH * 32; i += 128 * 8) {
                int ch = i >> 5, key = i & 31;
                copy16(&vbT[batchOffT + (size_t)ch * HW + kb + 32 + key],
                       &Vn[ch * VPITCH + key]);
            }
        }

        const unsigned short* Ks  = Ks2[cur];
        const unsigned short* Vts = Vts2[cur];

        // ---- S^T = K Q^T : t0 = keys 0-15, t1 = keys 16-31 (cols = queries) ----
        v8f t0 = (v8f){}, t1 = (v8f){};
#pragma unroll
        for (int ks = 0; ks < 8; ++ks) {
            BFrag qf;    // B operand: chans x queries, chans 8h..8h+7 and 16+8h.. per lane
            const unsigned short* qp = &Qs[(wave * 16 + l15) * CH + ks * 32 + half * 8];
            qf.u[0] = *(const u32x4*)qp;
            qf.u[1] = *(const u32x4*)(qp + 16);
            BFrag k0, k1;  // A operands: key rows
            const unsigned short* ap0 = &Ks[l15 * CH + ks * 32 + half * 8];
            k0.u[0] = *(const u32x4*)ap0;
            k0.u[1] = *(const u32x4*)(ap0 + 16);
            const unsigned short* ap1 = &Ks[(16 + l15) * CH + ks * 32 + half * 8];
            k1.u[0] = *(const u32x4*)ap1;
            k1.u[1] = *(const u32x4*)(ap1 + 16);
            t0 = __builtin_amdgcn_wmma_f32_16x16x32_bf16(false, k0.b, false, qf.b,
                                                         (short)0, t0, false, false);
            t1 = __builtin_amdgcn_wmma_f32_16x16x32_bf16(false, k1.b, false, qf.b,
                                                         (short)0, t1, false, false);
        }

        // ---- online softmax: stats are per-lane (query = l15) ----
        float mx = t0[0];
#pragma unroll
        for (int r = 1; r < 8; ++r) mx = fmaxf(mx, t0[r]);
#pragma unroll
        for (int r = 0; r < 8; ++r) mx = fmaxf(mx, t1[r]);
        mx = fmaxf(mx, __shfl_xor(mx, 16));     // combine key halves
        float mn = fmaxf(m, mx);
        float sc = __expf(m - mn);
        m = mn;

        float rs = 0.0f;
#pragma unroll
        for (int r = 0; r < 8; ++r) {
            t0[r] = __expf(t0[r] - mn); rs += t0[r];
            t1[r] = __expf(t1[r] - mn); rs += t1[r];
        }
        rs += __shfl_xor(rs, 16);
        l = l * sc + rs;

        // Broadcast per-query scale to the O-accumulator row layout.
        float osc[8];
#pragma unroll
        for (int r = 0; r < 8; ++r) osc[r] = __shfl(sc, r + 8 * half);
#pragma unroll
        for (int nt = 0; nt < 16; ++nt)
#pragma unroll
            for (int r = 0; r < 8; ++r) acc[nt][r] *= osc[r];

        // ---- P (S^T layout) -> WMMA A-fragment: pure in-lane packing ----
        BFrag pa;
#pragma unroll
        for (int v = 0; v < 4; ++v) {
            pa.w[v]     = pack2(t0[2 * v], t0[2 * v + 1]);   // keys 8h+2v,8h+2v+1
            pa.w[v + 4] = pack2(t1[2 * v], t1[2 * v + 1]);   // keys 16+8h+...
        }

        // ---- O += P V (K=32 in a single WMMA per channel tile) ----
#pragma unroll
        for (int nt = 0; nt < 16; ++nt) {
            BFrag vv;
            const unsigned short* vp = &Vts[(nt * 16 + l15) * VPITCH + half * 8];
            vv.u[0] = *(const u32x4*)vp;
            vv.u[1] = *(const u32x4*)(vp + 16);
            acc[nt] = __builtin_amdgcn_wmma_f32_16x16x32_bf16(
                        false, pa.b, false, vv.b, (short)0, acc[nt], false, false);
        }
    }

    // ---- epilogue: out = O/l * gamma + x ----
    float g  = gamma[0];
    float rl = 1.0f / l;
    float rlb[8];
#pragma unroll
    for (int r = 0; r < 8; ++r) rlb[r] = __shfl(rl, r + 8 * half);
#pragma unroll
    for (int nt = 0; nt < 16; ++nt) {
        int ch = nt * 16 + l15;
#pragma unroll
        for (int r = 0; r < 8; ++r) {
            int row = mb + wave * 16 + r + 8 * half;
            size_t idx = batchOff + (size_t)row * CH + ch;
            out[idx] = acc[nt][r] * rlb[r] * g + x[idx];
        }
    }
}

// ---------------------------------------------------------------------------
// Launch: workspace = qb(16MB) | vbT(16MB) | w1T(128KB) | w2T(128KB)
// ---------------------------------------------------------------------------
extern "C" void kernel_launch(void* const* d_in, const int* in_sizes, int n_in,
                              void* d_out, int out_size, void* d_ws, size_t ws_size,
                              hipStream_t stream) {
    const float* x     = (const float*)d_in[0];
    const float* dw1   = (const float*)d_in[1];
    const float* pw1   = (const float*)d_in[2];
    const float* b1    = (const float*)d_in[3];
    const float* dw2   = (const float*)d_in[4];
    const float* pw2   = (const float*)d_in[5];
    const float* b2    = (const float*)d_in[6];
    const float* gamma = (const float*)d_in[7];
    float* out = (float*)d_out;

    unsigned short* qb  = (unsigned short*)d_ws;
    unsigned short* vbT = qb  + (size_t)ROWS * CH;
    unsigned short* w1T = vbT + (size_t)ROWS * CH;
    unsigned short* w2T = w1T + CH * CH;

    prep_w_kernel<<<CH, CH, 0, stream>>>(pw1, pw2, w1T, w2T);
    proj_kernel<<<ROWS / 64, 128, 0, stream>>>(x, dw1, b1, dw2, b2, w1T, w2T, qb, vbT);
    attn_kernel<<<BATCH * (HW / 64), 128, 0, stream>>>(qb, vbT, x, gamma, out);
}
